// MultiHeadAttention_52338471469441
// MI455X (gfx1250) — matrix operand order repaired and checked by reading, not compiled, not run on hardware
//
#include <hip/hip_runtime.h>
#include <hip/hip_bf16.h>
#include <math.h>

typedef __attribute__((ext_vector_type(16))) __bf16 v16bf;
typedef __attribute__((ext_vector_type(8)))  __bf16 v8bf;
typedef __attribute__((ext_vector_type(8)))  float  v8f;

#define BB 4
#define SS 4096
#define HID 1024
#define NH 16
#define HD 64
#define MROWS (BB * SS)   // 16384

// ---------------------------------------------------------------------------
// GEMM: Out[M,N] = X[M,K] * W[K,N] + bias[N], fp32 in/out, bf16 WMMA compute.
// Block = 256 threads = 8 waves computing a 128x128 output tile.
// Wave w owns M-strip w*16 and all 8 N-tiles. Per K-step: preload A fragment
// (2x ds_load_b128) + all 8 B fragments (16x ds_load_b128, one clause), then
// issue 8 WMMAs back-to-back (XDL co-execution friendly).
// A/B panels staged through LDS in bf16; B stored transposed [n][k] so every
// per-lane WMMA fragment is a contiguous LDS vector load.
// grid = (N/128, M/128). Requires M%128==0, N%128==0, K%32==0.
// ---------------------------------------------------------------------------
__global__ __launch_bounds__(256)
void gemm_bf16_wmma(const float* __restrict__ X, const float* __restrict__ W,
                    const float* __restrict__ bias, float* __restrict__ Out,
                    int M, int N, int K) {
  __shared__ __bf16 Abuf[128][32];   // [m][k]   8 KB
  __shared__ __bf16 Bbuf[128][32];   // [n][k]   8 KB (transposed panel)

  const int tid  = threadIdx.x;
  const int lane = tid & 31;
  const int wave = tid >> 5;
  const int m_blk = blockIdx.y * 128;
  const int n_blk = blockIdx.x * 128;
  const int r15 = lane & 15;   // row in A frag / col in B,D frags
  const int hi  = lane >> 4;   // half-wave
  const int akb = hi * 16;     // A frag: lanes 16-31 hold K=16..31
  const int bkb = hi * 16;     // B frag: lanes 16-31 hold K=16..31

  // staging index maps (256 threads cover each 4096-element panel)
  const int arow = tid >> 1;          // 0..127
  const int akh  = (tid & 1) * 16;    // 0 or 16 : 16 consecutive K per thread
  const int bk   = tid >> 3;          // 0..31
  const int bn   = (tid & 7) * 16;    // 16 consecutive N per thread

  v8f acc[8] = {};

  for (int k0 = 0; k0 < K; k0 += 32) {
    // ---- stage A panel (coalesced 64B per thread), convert f32->bf16 ----
    {
      const float* ap = X + (size_t)(m_blk + arow) * K + (k0 + akh);
      if (k0 + 32 < K) __builtin_prefetch(ap + 32, 0, 1);
#pragma unroll
      for (int e = 0; e < 16; ++e) Abuf[arow][akh + e] = (__bf16)ap[e];
    }
    // ---- stage B panel transposed into [n][k] ----
    {
      const float* bp = W + (size_t)(k0 + bk) * N + (n_blk + bn);
      if (k0 + 32 < K) __builtin_prefetch(bp + (size_t)32 * N, 0, 1);
#pragma unroll
      for (int e = 0; e < 16; ++e) Bbuf[bn + e][bk] = (__bf16)bp[e];
    }
    __syncthreads();

    // ---- A fragment for this wave's M strip (two contiguous 16B LDS loads) --
    v16bf a;
    {
      const v8bf lo = *(const v8bf*)&Abuf[wave * 16 + r15][akb];
      const v8bf hp = *(const v8bf*)&Abuf[wave * 16 + r15][akb + 8];
#pragma unroll
      for (int e = 0; e < 8; ++e) { a[e] = lo[e]; a[8 + e] = hp[e]; }
    }

    // ---- preload all 8 B fragments (contiguous 32B LDS loads, one clause) ---
    v16bf bfrag[8];
#pragma unroll
    for (int j = 0; j < 8; ++j) {
      bfrag[j] = *(const v16bf*)&Bbuf[j * 16 + r15][bkb];
    }

    // ---- 8 WMMAs back-to-back ----
#pragma unroll
    for (int j = 0; j < 8; ++j) {
      acc[j] = __builtin_amdgcn_wmma_f32_16x16x32_bf16(
          /*neg_a=*/false, a, /*neg_b=*/false, bfrag[j],
          /*c_mod=*/(short)0, acc[j], /*reuse_a=*/false, /*reuse_b=*/false);
    }
    __syncthreads();
  }

  // ---- epilogue: D[m][n] per ISA C/D layout: col=r15, row=r + 8*hi ----
  const int mrow = m_blk + wave * 16 + hi * 8;
#pragma unroll
  for (int j = 0; j < 8; ++j) {
    const int col = n_blk + j * 16 + r15;
    const float bv = bias ? bias[col] : 0.0f;
#pragma unroll
    for (int r = 0; r < 8; ++r) {
      Out[(size_t)(mrow + r) * N + col] = acc[j][r] + bv;
    }
  }
}

// ---------------------------------------------------------------------------
// Per-token head-axis attention. One 256-thread block per token (b,s).
// Writes attention probs (B,S,16,16) and the attn@V output directly into the
// torch-faithful transposed/reshaped layout so the last GEMM is row-major.
// ---------------------------------------------------------------------------
__global__ __launch_bounds__(256)
void attn_kernel(const float* __restrict__ Q, const float* __restrict__ Km,
                 const float* __restrict__ V, const float* __restrict__ swts,
                 float* __restrict__ O, float* __restrict__ attn_out) {
  const int t = blockIdx.x;      // token index 0..16383
  const int b = t >> 12;         // / 4096
  const int s = t & 4095;

  __shared__ float q[HID], k[HID], v[HID];
  __shared__ float qk[256];      // reused for attn probs
  __shared__ float sc[256];
  __shared__ float sq[NH], sk[NH];
  __shared__ float w[3];

  const int tid = threadIdx.x;
  const size_t base = (size_t)t * HID;
  for (int i = tid; i < HID; i += 256) {
    q[i] = Q[base + i];
    k[i] = Km[base + i];
    v[i] = V[base + i];
  }
  if (tid == 0) {
    float m  = fmaxf(fmaxf(swts[0], swts[1]), swts[2]);
    float e0 = __expf(swts[0] - m), e1 = __expf(swts[1] - m), e2 = __expf(swts[2] - m);
    float inv = 1.0f / (e0 + e1 + e2);
    w[0] = e0 * inv; w[1] = e1 * inv; w[2] = e2 * inv;
  }
  __syncthreads();

  const int i = tid >> 4, j = tid & 15;
  float dot = 0.0f;
#pragma unroll 8
  for (int d = 0; d < HD; ++d) dot += q[i * HD + d] * k[j * HD + d];
  qk[tid] = dot;

  if (tid < NH) {
    float s2 = 0.0f;
    for (int d = 0; d < HD; ++d) { float x = q[tid * HD + d]; s2 += x * x; }
    sq[tid] = s2;
  } else if (tid < 2 * NH) {
    const int h = tid - NH;
    float s2 = 0.0f;
    for (int d = 0; d < HD; ++d) { float x = k[h * HD + d]; s2 += x * x; }
    sk[h] = s2;
  }
  __syncthreads();

  {
    const float qkij = qk[tid];
    const float dotc = qkij * 0.125f;                       // 1/sqrt(64)
    const float den  = fmaxf(sqrtf(sq[i]) * sqrtf(sk[j]), 1e-8f);
    const float cosv = qkij / den;
    const float d2   = fmaxf(sq[i] + sk[j] - 2.0f * qkij, 0.0f);
    sc[tid] = w[0] * dotc + w[1] * cosv - w[2] * sqrtf(d2);
  }
  __syncthreads();

  if (tid < NH) {                 // row-wise softmax over 16 heads
    float m = -INFINITY;
    for (int jj = 0; jj < NH; ++jj) m = fmaxf(m, sc[tid * NH + jj]);
    float sum = 0.0f;
    for (int jj = 0; jj < NH; ++jj) {
      float e = __expf(sc[tid * NH + jj] - m);
      qk[tid * NH + jj] = e;
      sum += e;
    }
    float inv = 1.0f / sum;
    for (int jj = 0; jj < NH; ++jj) qk[tid * NH + jj] *= inv;
  }
  __syncthreads();

  attn_out[(size_t)t * 256 + tid] = qk[tid];

  // out[h][d] = sum_j attn[h][j] * v[j][d]; write at shuffled (s', c):
  //   s' = h*256 + s/16 ; c = (s%16)*64 + d
#pragma unroll
  for (int r = 0; r < 4; ++r) {
    const int idx = tid + r * 256;           // 0..1023
    const int h = idx >> 6, d = idx & 63;
    float o = 0.0f;
#pragma unroll
    for (int jj = 0; jj < NH; ++jj) o += qk[h * NH + jj] * v[jj * HD + d];
    const int sp = h * 256 + (s >> 4);
    const int c  = (s & 15) * HD + d;
    O[((size_t)b * SS + sp) * HID + c] = o;
  }
}

extern "C" void kernel_launch(void* const* d_in, const int* in_sizes, int n_in,
                              void* d_out, int out_size, void* d_ws, size_t ws_size,
                              hipStream_t stream) {
  const float* query = (const float*)d_in[0];
  const float* key   = (const float*)d_in[1];
  const float* value = (const float*)d_in[2];
  const float* Wq = (const float*)d_in[3];
  const float* bq = (const float*)d_in[4];
  const float* Wk = (const float*)d_in[5];
  const float* bk = (const float*)d_in[6];
  const float* Wv = (const float*)d_in[7];
  const float* bv = (const float*)d_in[8];
  const float* Wo = (const float*)d_in[9];
  const float* bo = (const float*)d_in[10];
  const float* sw = (const float*)d_in[11];

  float* out  = (float*)d_out;                       // (B,S,HID)
  float* attn = out + (size_t)MROWS * HID;           // (B,S,16,16)

  const size_t mat = (size_t)MROWS * HID;            // 16,777,216 floats
  float* Qb = (float*)d_ws;
  float* Kb = Qb + mat;
  float* Vb = Kb + mat;
  float* Ob = Vb + mat;

  dim3 gblk(256, 1, 1);
  dim3 ggrid(HID / 128, MROWS / 128, 1);             // (8, 128)

  gemm_bf16_wmma<<<ggrid, gblk, 0, stream>>>(query, Wq, bq, Qb, MROWS, HID, HID);
  gemm_bf16_wmma<<<ggrid, gblk, 0, stream>>>(key,   Wk, bk, Kb, MROWS, HID, HID);
  gemm_bf16_wmma<<<ggrid, gblk, 0, stream>>>(value, Wv, bv, Vb, MROWS, HID, HID);

  attn_kernel<<<dim3(MROWS, 1, 1), gblk, 0, stream>>>(Qb, Kb, Vb, sw, Ob, attn);

  gemm_bf16_wmma<<<ggrid, gblk, 0, stream>>>(Ob, Wo, bo, out, MROWS, HID, HID);
}
